// JANETCell_85289460564575
// MI455X (gfx1250) — matrix operand order repaired and checked by reading, not compiled, not run on hardware
//
#include <hip/hip_runtime.h>
#include <hip/hip_bf16.h>

// ---------------------------------------------------------------------------
// JANET cell, B=128 T=1024 D=256 U=256, fp32 in/out.
// Phase 0: convert kernel / recurrent_kernel to bf16 WMMA-B fragment layout.
// Phase 1: xz = inputs @ kernel + bias  (bf16 WMMA, fp32 accum) -> ws
// Phase 2: sequential scan; 8 blocks (one per 16 batch rows), 16 waves/block
//          (one per 16-col tile of U); h AND recurrent weights live in LDS.
// ---------------------------------------------------------------------------

typedef __attribute__((ext_vector_type(16))) __bf16 v16bf;
typedef __attribute__((ext_vector_type(8)))  __bf16 v8bf;
typedef __attribute__((ext_vector_type(8)))  float  v8f;
typedef __attribute__((ext_vector_type(4)))  float  v4f;

#define B_   128
#define T_   1024
#define D_   256
#define U_   256
#define NU2  512              // 2*U

#define HB_ELEMS   (2 * 16 * 264)        // double-buffered h, padded rows
#define HB_BYTES   (HB_ELEMS * 2)        // 16896 B
#define WR_ELEMS   (2 * 16 * 8 * 512)    // 131072 bf16 fragments
#define WR_BYTES   (WR_ELEMS * 2)        // 262144 B
#define SMEM_BYTES (HB_BYTES + WR_BYTES) // 279040 B  (< 320KB WGP LDS)

__device__ __forceinline__ v8f wmma_bf16(v16bf a, v16bf b, v8f c) {
  // (neg_a, A, neg_b, B, c_mod, C, reuse_a, reuse_b)
  return __builtin_amdgcn_wmma_f32_16x16x32_bf16(false, a, false, b,
                                                 (short)0, c, false, false);
}

// ---------------------------------------------------------------------------
// Phase 0: pack fp32 weights into bf16 WMMA-B fragments.
// Fragment f = (n_tile*8 + k_tile), 512 bf16 each, stored as:
//   index = f*512 + half*256 + lane*8 + j      (half: VGPRs 0-3 vs 4-7)
// B-operand layout (32x16, K x N): lane -> n = lane%16;
//   k = k_tile*32 + half*16 + (lane/16)*8 + j
// ---------------------------------------------------------------------------
__global__ __launch_bounds__(256) void prep_weights(
    const float* __restrict__ kernel, const float* __restrict__ rkern,
    __bf16* __restrict__ wk, __bf16* __restrict__ wr) {
  int tid = blockIdx.x * 256 + threadIdx.x;      // 0 .. 262143
  const float* src;
  __bf16* dst;
  int e;
  if (tid < 131072) { e = tid;          src = kernel; dst = wk; }
  else              { e = tid - 131072; src = rkern;  dst = wr; }
  int f  = e >> 9;
  int r  = e & 511;
  int h2 = r >> 8;
  int L  = (r >> 3) & 31;
  int j  = r & 7;
  int kt = f & 7;
  int nt = f >> 3;
  int lo = L & 15, hi = L >> 4;
  int k   = kt * 32 + h2 * 16 + hi * 8 + j;
  int col = nt * 16 + lo;
  dst[e] = (__bf16)src[k * NU2 + col];
}

// ---------------------------------------------------------------------------
// Phase 1: xz[m][n] = inputs[m][:] @ kernel[:, n] + bias[n]
// M = B*T = 131072, N = 512, K = 256.  One wave per 16x16 output tile.
// ---------------------------------------------------------------------------
__global__ __launch_bounds__(512) void xz_gemm(
    const float* __restrict__ in, const __bf16* __restrict__ wk,
    const float* __restrict__ bias, float* __restrict__ xz) {
  int tid  = threadIdx.x;
  int wave = tid >> 5, lane = tid & 31;
  int lo = lane & 15, hi = lane >> 4;
  int w = blockIdx.x * 16 + wave;
  int row_tile = w >> 5;        // 8192 row tiles
  int col_tile = w & 31;        // 32 col tiles
  int m0 = row_tile << 4;

  const float* arow = in + (size_t)(m0 + lo) * D_;
  v8f acc = {};
#pragma unroll
  for (int kt = 0; kt < 8; ++kt) {
    int kb = kt * 32 + hi * 8;
    v4f a0 = *(const v4f*)(arow + kb);
    v4f a1 = *(const v4f*)(arow + kb + 4);
    v4f a2 = *(const v4f*)(arow + kb + 16);
    v4f a3 = *(const v4f*)(arow + kb + 20);
    v16bf a;
#pragma unroll
    for (int i = 0; i < 4; ++i) {
      a[i]      = (__bf16)a0[i];
      a[4 + i]  = (__bf16)a1[i];
      a[8 + i]  = (__bf16)a2[i];
      a[12 + i] = (__bf16)a3[i];
    }
    const __bf16* bp = wk + (((size_t)col_tile * 8 + kt) << 9);
    v8bf b0 = *(const v8bf*)(bp + lane * 8);
    v8bf b1 = *(const v8bf*)(bp + 256 + lane * 8);
    v16bf b;
#pragma unroll
    for (int i = 0; i < 8; ++i) { b[i] = b0[i]; b[8 + i] = b1[i]; }
    acc = wmma_bf16(a, b, acc);
  }
  float bv = bias[col_tile * 16 + lo];
#pragma unroll
  for (int r = 0; r < 8; ++r)   // D layout: row = r + hi*8, col = lo
    xz[(size_t)(m0 + hi * 8 + r) * NU2 + col_tile * 16 + lo] = acc[r] + bv;
}

// ---------------------------------------------------------------------------
// Phase 2: sequential scan.  grid = 8 blocks, 512 thr = 16 waves.
// Dynamic LDS layout: [0, 16896) = h double buffer (bf16, rows padded to 264)
//                     [16896, 279040) = recurrent-weight bf16 fragments.
// Per step/wave: 16 WMMAs fed entirely from LDS, fused sigmoid/tanh cell
// update, non-temporal fp32 h -> global, bf16 h -> LDS, one barrier.
// Next step's xz lines are prefetched (global_prefetch_b8) during the WMMAs.
// ---------------------------------------------------------------------------
__global__ __launch_bounds__(512) void janet_scan(
    const float* __restrict__ xz, const __bf16* __restrict__ wr,
    const float* __restrict__ h0, const float* __restrict__ c0,
    float* __restrict__ out) {
  extern __shared__ __align__(16) char smem[];
  __bf16* hb   = (__bf16*)smem;                 // [2][16][264]
  __bf16* wlds = (__bf16*)(smem + HB_BYTES);    // 131072 bf16 fragments

  int tid  = threadIdx.x;
  int n    = tid >> 5;          // wave id == column tile
  int lane = tid & 31;
  int lo = lane & 15, hi = lane >> 4;
  int b0  = blockIdx.x << 4;    // batch row base
  int col = n * 16 + lo;

  // ---- preload recurrent-weight fragments into LDS (b128 copies) ----
  {
    const uint4* s = (const uint4*)wr;
    uint4*       d = (uint4*)wlds;
#pragma unroll 4
    for (int i = tid; i < WR_BYTES / 16; i += 512) d[i] = s[i];
  }

  // ---- init c state (regs) and h0 (LDS buffer 0) ----
  float cst[8];
#pragma unroll
  for (int r = 0; r < 8; ++r) {
    int m = hi * 8 + r;
    cst[r] = c0[(size_t)(b0 + m) * U_ + col];
    hb[m * 264 + col] = (__bf16)h0[(size_t)(b0 + m) * U_ + col];
  }
  __syncthreads();

  const __bf16* wf = wlds + ((size_t)(n)      * 8 << 9);   // f-gate frags
  const __bf16* wc = wlds + ((size_t)(16 + n) * 8 << 9);   // candidate frags

  for (int t = 0; t < T_; ++t) {
    int cur = t & 1, nxt = cur ^ 1;
    const __bf16* ha = hb + cur * (16 * 264);
    __bf16*       hw = hb + nxt * (16 * 264);

    // pre-activations for this step (issued early, consumed after WMMAs)
    float xf[8], xc[8];
#pragma unroll
    for (int r = 0; r < 8; ++r) {
      size_t rb = ((size_t)(b0 + hi * 8 + r) * T_ + t) * NU2;
      xf[r] = xz[rb + col];
      xc[r] = xz[rb + U_ + col];
    }

    // prefetch next step's xz lines while this step's WMMAs run
    if (t + 1 < T_) {
      if (lo == 0) {
#pragma unroll
        for (int r = 0; r < 8; ++r) {
          size_t rb = ((size_t)(b0 + hi * 8 + r) * T_ + (t + 1)) * NU2;
          __builtin_prefetch(&xz[rb + col], 0, 3);
          __builtin_prefetch(&xz[rb + U_ + col], 0, 3);
        }
      }
    }

    v8f accf = {}, accc = {};
#pragma unroll
    for (int kt = 0; kt < 8; ++kt) {
      int kb = kt * 32 + hi * 8;
      v8bf a0 = *(const v8bf*)&ha[lo * 264 + kb];
      v8bf a1 = *(const v8bf*)&ha[lo * 264 + kb + 16];
      v16bf a;
#pragma unroll
      for (int i = 0; i < 8; ++i) { a[i] = a0[i]; a[8 + i] = a1[i]; }

      const __bf16* pf = wf + (kt << 9);
      v8bf f0 = *(const v8bf*)(pf + lane * 8);
      v8bf f1 = *(const v8bf*)(pf + 256 + lane * 8);
      v16bf bfm;
#pragma unroll
      for (int i = 0; i < 8; ++i) { bfm[i] = f0[i]; bfm[8 + i] = f1[i]; }
      accf = wmma_bf16(a, bfm, accf);

      const __bf16* pc = wc + (kt << 9);
      v8bf g0 = *(const v8bf*)(pc + lane * 8);
      v8bf g1 = *(const v8bf*)(pc + 256 + lane * 8);
      v16bf bcm;
#pragma unroll
      for (int i = 0; i < 8; ++i) { bcm[i] = g0[i]; bcm[8 + i] = g1[i]; }
      accc = wmma_bf16(a, bcm, accc);
    }

#pragma unroll
    for (int r = 0; r < 8; ++r) {
      int m = hi * 8 + r;
      float fg   = 1.0f / (1.0f + __expf(-(xf[r] + accf[r])));
      float cand = tanhf(xc[r] + accc[r]);
      float cn   = fg * cst[r] + (1.0f - fg) * cand;
      cst[r] = cn;
      // h stream is written once and never re-read: keep it out of L2
      __builtin_nontemporal_store(cn, &out[((size_t)(b0 + m) * T_ + t) * U_ + col]);
      hw[m * 264 + col] = (__bf16)cn;
    }
    __syncthreads();
  }
}

// ---------------------------------------------------------------------------
extern "C" void kernel_launch(void* const* d_in, const int* in_sizes, int n_in,
                              void* d_out, int out_size, void* d_ws, size_t ws_size,
                              hipStream_t stream) {
  const float* inputs = (const float*)d_in[0];   // (128,1024,256)
  const float* h0     = (const float*)d_in[1];   // (128,256)
  const float* c0     = (const float*)d_in[2];   // (128,256)
  const float* kern   = (const float*)d_in[3];   // (256,512)
  const float* rkern  = (const float*)d_in[4];   // (256,512)
  const float* bias   = (const float*)d_in[5];   // (512,)
  float* out = (float*)d_out;                    // (128,1024,256)

  char* ws = (char*)d_ws;
  const size_t XZ_BYTES = (size_t)B_ * T_ * NU2 * sizeof(float);  // 268 MB
  const size_t WF_BYTES = (size_t)D_ * NU2 * sizeof(__bf16);      // 256 KB
  float*  xz = (float*)ws;
  __bf16* wk = (__bf16*)(ws + XZ_BYTES);
  __bf16* wr = (__bf16*)(ws + XZ_BYTES + WF_BYTES);

  // allow 279040 B of dynamic LDS for the scan kernel (deterministic, host-side)
  (void)hipFuncSetAttribute((const void*)janet_scan,
                            hipFuncAttributeMaxDynamicSharedMemorySize,
                            SMEM_BYTES);

  // Phase 0: 262144 bf16 elements -> 1024 blocks x 256
  prep_weights<<<1024, 256, 0, stream>>>(kern, rkern, wk, wr);
  // Phase 1: (131072/16)*(512/16) tiles / 16 waves = 16384 blocks
  xz_gemm<<<16384, 512, 0, stream>>>(inputs, wk, bias, xz);
  // Phase 2: 8 batch tiles, 16 waves each, 279040 B dynamic LDS
  janet_scan<<<8, 512, SMEM_BYTES, stream>>>(xz, wr, h0, c0, out);
}
